// GroupBRouter_78288663872362
// MI455X (gfx1250) — matrix operand order repaired and checked by reading, not compiled, not run on hardware
//
#include <hip/hip_runtime.h>
#include <hip/hip_bf16.h>
#include <math.h>

// ---------------------------------------------------------------------------
// Shapes (compile-time constants from the reference)
// ---------------------------------------------------------------------------
#define BB   4
#define NBQ  2048
#define NKV  2048
#define DD   1024
#define HH   16
#define HDQ  64
#define EE   8
#define MROWS (BB * NBQ)   // 8192

typedef __attribute__((ext_vector_type(16))) __bf16 v16bf;
typedef __attribute__((ext_vector_type(8)))  float  v8f;

union Frag { unsigned int u[8]; v16bf v; };

__device__ inline v8f wmma_bf16(Frag a, Frag b, v8f c) {
  return __builtin_amdgcn_wmma_f32_16x16x32_bf16(false, a.v, false, b.v,
                                                 (short)0, c, false, false);
}

// ---------------------------------------------------------------------------
// CDNA5 async global->LDS DMA (tracked by ASYNCcnt). LDS address = low 32
// bits of the generic shared-memory pointer (ISA: LDS aperture uses
// addr[31:0]). GV addressing mode: 64-bit vaddr, saddr = off.
// ---------------------------------------------------------------------------
__device__ __forceinline__ void async_ld_b128(void* lds, const void* gmem) {
  unsigned l = (unsigned)(unsigned long long)lds;
  unsigned long long g = (unsigned long long)gmem;
  asm volatile("global_load_async_to_lds_b128 %0, %1, off"
               :: "v"(l), "v"(g) : "memory");
}
__device__ __forceinline__ void wait_async0() {
  asm volatile("s_wait_asynccnt 0x0" ::: "memory");
}
__device__ __forceinline__ void wait_async4() {
  asm volatile("s_wait_asynccnt 0x4" ::: "memory");
}

// ---------------------------------------------------------------------------
// f32 -> bf16 conversion kernels
// ---------------------------------------------------------------------------
__global__ void k_f2bf(const float* __restrict__ in, __bf16* __restrict__ out, int n) {
  for (int i = blockIdx.x * blockDim.x + threadIdx.x; i < n; i += gridDim.x * blockDim.x)
    out[i] = (__bf16)in[i];
}

// ctx_AC = concat(output_A, output_C) along axis 1, converted to bf16
__global__ void k_ctx_bf(const float* __restrict__ A, const float* __restrict__ C,
                         __bf16* __restrict__ out) {
  const int total = BB * NKV * DD;
  for (int i = blockIdx.x * blockDim.x + threadIdx.x; i < total; i += gridDim.x * blockDim.x) {
    int b   = i / (NKV * DD);
    int rem = i - b * (NKV * DD);
    int r   = rem >> 10;
    int d   = rem & 1023;
    float v = (r < 1024) ? A[((size_t)b * 1024 + r) * DD + d]
                         : C[((size_t)b * 1024 + (r - 1024)) * DD + d];
    out[i] = (__bf16)v;
  }
}

// ---------------------------------------------------------------------------
// Tiled bf16 GEMM:  C[M,N] = A[M,K] @ W[N,K]^T + bias[N]
// mode 0: f32 out,  mode 1: bf16 out,  mode 2: bf16 out + ReLU
// 256 threads (8 waves), block tile 128x128, wave tile 64x32.
// Double-buffered LDS slabs filled by async global->LDS DMA; each thread
// issues 4 async b128 ops per slab (in-order completion per wave), so
// s_wait_asynccnt<=4 after issuing slab i+1 means slab i has landed.
// ---------------------------------------------------------------------------
__global__ __launch_bounds__(256)
void k_gemm(const __bf16* __restrict__ A, const __bf16* __restrict__ W,
            const float* __restrict__ bias,
            float* __restrict__ outF, __bf16* __restrict__ outH,
            int M, int N, int K, int mode) {
  __shared__ unsigned short As[2][128 * 32];
  __shared__ unsigned short Ws[2][128 * 32];

  const int tid  = threadIdx.x;
  const int lane = tid & 31;
  const int w    = tid >> 5;
  const int g    = lane >> 4;   // half-wave group
  const int lr16 = lane & 15;

  const int mBase = blockIdx.y * 128;
  const int nBase = blockIdx.x * 128;
  const int wm = (w >> 2) * 64;   // 0 / 64
  const int wn = (w & 3) * 32;    // 0..96

  v8f acc[4][2];
#pragma unroll
  for (int mt = 0; mt < 4; ++mt)
#pragma unroll
    for (int nt = 0; nt < 2; ++nt)
#pragma unroll
      for (int r = 0; r < 8; ++r) acc[mt][nt][r] = 0.0f;

  auto stage = [&](int buf, int k0) {
#pragma unroll
    for (int it = 0; it < 2; ++it) {
      int i = tid + it * 256;
      int row = i >> 2, ch = i & 3;
      async_ld_b128(&As[buf][row * 32 + ch * 8],
                    &A[(size_t)(mBase + row) * K + k0 + ch * 8]);
      async_ld_b128(&Ws[buf][row * 32 + ch * 8],
                    &W[(size_t)(nBase + row) * K + k0 + ch * 8]);
    }
  };

  const int nslab = K >> 5;
  stage(0, 0);
  for (int s2 = 0; s2 < nslab; ++s2) {
    const int cur = s2 & 1;
    if (s2 + 1 < nslab) {            // uniform branch: barrier-safe
      stage(cur ^ 1, (s2 + 1) << 5); // prefetch next slab (async DMA)
      wait_async4();                 // oldest 4 (slab s2) have landed
    } else {
      wait_async0();
    }
    __syncthreads();

    Frag af[4], bf[2];
#pragma unroll
    for (int mt = 0; mt < 4; ++mt) {
      int row = wm + mt * 16 + lr16;
#pragma unroll
      for (int p = 0; p < 8; ++p) {
        int k = g * 16 + 2 * p;
        af[mt].u[p] = *(const unsigned int*)&As[cur][row * 32 + k];
      }
    }
#pragma unroll
    for (int nt = 0; nt < 2; ++nt) {
      int nrow = wn + nt * 16 + lr16;
#pragma unroll
      for (int p = 0; p < 8; ++p) {
        int k = g * 16 + 2 * p;
        bf[nt].u[p] = *(const unsigned int*)&Ws[cur][nrow * 32 + k];
      }
    }
#pragma unroll
    for (int mt = 0; mt < 4; ++mt)
#pragma unroll
      for (int nt = 0; nt < 2; ++nt)
        acc[mt][nt] = wmma_bf16(af[mt], bf[nt], acc[mt][nt]);
    __syncthreads();
  }

  // --- epilogue ---
#pragma unroll
  for (int mt = 0; mt < 4; ++mt) {
#pragma unroll
    for (int nt = 0; nt < 2; ++nt) {
      int col = nBase + wn + nt * 16 + lr16;
      float bv = bias ? bias[col] : 0.0f;
#pragma unroll
      for (int r = 0; r < 8; ++r) {
        int row = mBase + wm + mt * 16 + r + 8 * g;
        float v = acc[mt][nt][r] + bv;
        if (mode == 2) v = fmaxf(v, 0.0f);
        if (mode == 0) outF[(size_t)row * N + col] = v;
        else           outH[(size_t)row * N + col] = (__bf16)v;
      }
    }
  }
}

// ---------------------------------------------------------------------------
// Flash attention. Grid: (B*H, NB/64). Block: 128 threads = 4 waves.
// Each wave owns 16 query rows; 32-key slabs. Q/K tiles staged via async
// global->LDS DMA (overlaps the V transpose-scatter done with plain loads).
// ---------------------------------------------------------------------------
__global__ __launch_bounds__(128)
void k_attn(const __bf16* __restrict__ Q, const __bf16* __restrict__ Km,
            const __bf16* __restrict__ Vm, __bf16* __restrict__ O) {
  const int bh = blockIdx.x;
  const int b  = bh >> 4;
  const int h  = bh & 15;
  const int q0 = blockIdx.y * 64;

  __shared__ unsigned short Qs[64 * 64];   // 64 q rows x 64 d
  __shared__ unsigned short Ks[32 * 64];   // 32 keys  x 64 d
  __shared__ unsigned short Vst[64 * 32];  // 64 d x 32 keys (transposed)
  __shared__ unsigned short Ps[4][16 * 32];

  const int tid  = threadIdx.x;
  const int lane = tid & 31;
  const int w    = tid >> 5;
  const int g    = lane >> 4;
  const int lr16 = lane & 15;

  const __bf16* Qbase = Q + ((size_t)(b * NBQ + q0)) * DD + h * HDQ;
  for (int i = tid; i < 512; i += 128) {
    int row = i >> 3, ch = i & 7;
    async_ld_b128(&Qs[row * 64 + ch * 8], &Qbase[(size_t)row * DD + ch * 8]);
  }

  v8f o[4];
#pragma unroll
  for (int nt = 0; nt < 4; ++nt)
#pragma unroll
    for (int r = 0; r < 8; ++r) o[nt][r] = 0.0f;
  float mrun[8], lsum[8];
#pragma unroll
  for (int r = 0; r < 8; ++r) { mrun[r] = -1e30f; lsum[r] = 0.0f; }

  wait_async0();
  __syncthreads();

  for (int kt = 0; kt < NKV; kt += 32) {
    const __bf16* Kbase = Km + ((size_t)(b * NKV + kt)) * DD + h * HDQ;
    const __bf16* Vbase = Vm + ((size_t)(b * NKV + kt)) * DD + h * HDQ;
    // K slab: async DMA straight to LDS
    for (int i = tid; i < 256; i += 128) {
      int row = i >> 3, ch = i & 7;
      async_ld_b128(&Ks[row * 64 + ch * 8], &Kbase[(size_t)row * DD + ch * 8]);
    }
    // V slab: transpose-scatter (overlaps the K DMA)
    for (int i = tid; i < 256; i += 128) {
      int row = i >> 3, ch = i & 7;
      uint4 t4 = *(const uint4*)&Vbase[(size_t)row * DD + ch * 8];
      unsigned short* ts = (unsigned short*)&t4;
#pragma unroll
      for (int e = 0; e < 8; ++e) Vst[(ch * 8 + e) * 32 + row] = ts[e];
    }
    wait_async0();
    __syncthreads();

    // S = Q K^T (16 q x 32 keys per wave)
    v8f s[2];
#pragma unroll
    for (int nt2 = 0; nt2 < 2; ++nt2)
#pragma unroll
      for (int r = 0; r < 8; ++r) s[nt2][r] = 0.0f;
#pragma unroll
    for (int nt2 = 0; nt2 < 2; ++nt2) {
#pragma unroll
      for (int kk = 0; kk < 2; ++kk) {
        Frag aq, bk;
#pragma unroll
        for (int p = 0; p < 8; ++p) {
          int ka = kk * 32 + g * 16 + 2 * p;
          aq.u[p] = *(const unsigned int*)&Qs[(w * 16 + lr16) * 64 + ka];
          int kb = kk * 32 + g * 16 + 2 * p;
          bk.u[p] = *(const unsigned int*)&Ks[(nt2 * 16 + lr16) * 64 + kb];
        }
        s[nt2] = wmma_bf16(aq, bk, s[nt2]);
      }
    }

    // online softmax (rows = r + 8*g, cols spread over 16 lanes)
    const float scale = 0.125f;  // 1/sqrt(64)
#pragma unroll
    for (int r = 0; r < 8; ++r) {
      float s0 = s[0][r] * scale, s1 = s[1][r] * scale;
      float mx = fmaxf(s0, s1);
      mx = fmaxf(mx, __shfl_xor(mx, 1, 16));
      mx = fmaxf(mx, __shfl_xor(mx, 2, 16));
      mx = fmaxf(mx, __shfl_xor(mx, 4, 16));
      mx = fmaxf(mx, __shfl_xor(mx, 8, 16));
      float mnew = fmaxf(mrun[r], mx);
      float ef = __expf(mrun[r] - mnew);
      float p0 = __expf(s0 - mnew), p1 = __expf(s1 - mnew);
      float rs = p0 + p1;
      rs += __shfl_xor(rs, 1, 16);
      rs += __shfl_xor(rs, 2, 16);
      rs += __shfl_xor(rs, 4, 16);
      rs += __shfl_xor(rs, 8, 16);
      lsum[r] = lsum[r] * ef + rs;
      mrun[r] = mnew;
      o[0][r] *= ef; o[1][r] *= ef; o[2][r] *= ef; o[3][r] *= ef;
      int prow = (r + 8 * g) * 32;
      __bf16 h0 = (__bf16)p0, h1 = (__bf16)p1;
      Ps[w][prow + lr16]      = __builtin_bit_cast(unsigned short, h0);
      Ps[w][prow + 16 + lr16] = __builtin_bit_cast(unsigned short, h1);
    }
    __syncthreads();

    // O += P V
    Frag ap;
#pragma unroll
    for (int p = 0; p < 8; ++p) {
      int k = g * 16 + 2 * p;
      ap.u[p] = *(const unsigned int*)&Ps[w][lr16 * 32 + k];
    }
#pragma unroll
    for (int nt = 0; nt < 4; ++nt) {
      Frag bv;
#pragma unroll
      for (int p = 0; p < 8; ++p) {
        int kk2 = g * 16 + 2 * p;
        bv.u[p] = *(const unsigned int*)&Vst[(nt * 16 + lr16) * 32 + kk2];
      }
      o[nt] = wmma_bf16(ap, bv, o[nt]);
    }
    __syncthreads();
  }

  // normalize + store bf16
#pragma unroll
  for (int r = 0; r < 8; ++r) {
    float inv = 1.0f / lsum[r];
    int grow = b * NBQ + q0 + w * 16 + r + 8 * g;
    size_t base = (size_t)grow * DD + h * HDQ;
#pragma unroll
    for (int nt = 0; nt < 4; ++nt)
      O[base + nt * 16 + lr16] = (__bf16)(o[nt][r] * inv);
  }
}

// ---------------------------------------------------------------------------
// ctx_B = LayerNorm(proj + tokens). One block per row.
// ---------------------------------------------------------------------------
__global__ __launch_bounds__(256)
void k_ln_residual(const float* __restrict__ proj, const float* __restrict__ tokens,
                   const float* __restrict__ gamma, const float* __restrict__ beta,
                   float* __restrict__ out) {
  const int row = blockIdx.x;
  const int tid = threadIdx.x;
  __shared__ float red[256];
  const size_t base = (size_t)row * DD;
  float x[4];
  float s = 0.0f;
#pragma unroll
  for (int j = 0; j < 4; ++j) {
    x[j] = proj[base + tid + j * 256] + tokens[base + tid + j * 256];
    s += x[j];
  }
  red[tid] = s; __syncthreads();
  for (int off = 128; off > 0; off >>= 1) {
    if (tid < off) red[tid] += red[tid + off];
    __syncthreads();
  }
  float mu = red[0] / (float)DD; __syncthreads();
  float v = 0.0f;
#pragma unroll
  for (int j = 0; j < 4; ++j) { float d = x[j] - mu; v += d * d; }
  red[tid] = v; __syncthreads();
  for (int off = 128; off > 0; off >>= 1) {
    if (tid < off) red[tid] += red[tid + off];
    __syncthreads();
  }
  float var = red[0] / (float)DD;
  float rs = rsqrtf(var + 1e-5f);
#pragma unroll
  for (int j = 0; j < 4; ++j) {
    int c = tid + j * 256;
    out[base + c] = (x[j] - mu) * rs * gamma[c] + beta[c];
  }
}

// ---------------------------------------------------------------------------
// Time embedding + 2-layer MLP (tiny, VALU)
// ---------------------------------------------------------------------------
__global__ void k_time_embed(const int* __restrict__ t, float* __restrict__ emb) {
  int b = blockIdx.x;
  float tv = (float)t[b];
  for (int c = threadIdx.x; c < DD; c += blockDim.x) {
    float e;
    if (c < 512) { float f = __expf(-logf(10000.0f) * (float)c / 511.0f);        e = sinf(tv * f); }
    else         { float f = __expf(-logf(10000.0f) * (float)(c - 512) / 511.0f); e = cosf(tv * f); }
    emb[b * DD + c] = e;
  }
}

__global__ __launch_bounds__(256)
void k_te1(const float* __restrict__ emb, const float* __restrict__ w1,
           const float* __restrict__ b1, float* __restrict__ h1) {
  int idx = blockIdx.x * 256 + threadIdx.x;
  if (idx >= BB * 2048) return;
  int b = idx >> 11, j = idx & 2047;
  const float* e  = emb + b * DD;
  const float* wr = w1 + (size_t)j * DD;
  float s = b1[j];
  for (int i = 0; i < DD; ++i) s += e[i] * wr[i];
  h1[idx] = s / (1.0f + __expf(-s));  // SiLU
}

__global__ __launch_bounds__(256)
void k_te2(const float* __restrict__ h1, const float* __restrict__ w2,
           const float* __restrict__ b2, float* __restrict__ temb) {
  int idx = blockIdx.x * 256 + threadIdx.x;
  if (idx >= BB * DD) return;
  int b = idx >> 10, j = idx & 1023;
  const float* hr = h1 + b * 2048;
  const float* wr = w2 + (size_t)j * 2048;
  float s = b2[j];
  for (int i = 0; i < 2048; ++i) s += hr[i] * wr[i];
  temb[idx] = s;
}

// ---------------------------------------------------------------------------
// gate_in = concat(tokens, ctx_B, t_emb[b] broadcast) -> bf16
// ---------------------------------------------------------------------------
__global__ void k_gate_in(const float* __restrict__ tokens, const float* __restrict__ ctxB,
                          const float* __restrict__ temb, __bf16* __restrict__ gin, int total) {
  for (int idx = blockIdx.x * blockDim.x + threadIdx.x; idx < total;
       idx += gridDim.x * blockDim.x) {
    int row = idx / 3072;
    int c = idx - row * 3072;
    int b = row >> 11;
    float v;
    if (c < 1024)      v = tokens[(size_t)row * DD + c];
    else if (c < 2048) v = ctxB[(size_t)row * DD + (c - 1024)];
    else               v = temb[b * DD + (c - 2048)];
    gin[idx] = (__bf16)v;
  }
}

// gate layer 2: logits[row, e] = relu_hidden . w2[e] + b2[e] + expert_bias[e]
__global__ __launch_bounds__(256)
void k_gate2(const __bf16* __restrict__ hmat, const float* __restrict__ w2,
             const float* __restrict__ b2, const float* __restrict__ ebias,
             float* __restrict__ logits) {
  int idx = blockIdx.x * 256 + threadIdx.x;
  if (idx >= MROWS * EE) return;
  int row = idx >> 3, e = idx & 7;
  const __bf16* hr = hmat + (size_t)row * DD;
  const float*  wr = w2 + e * DD;
  float s = b2[e] + ebias[e];
  for (int i = 0; i < DD; ++i) s += (float)hr[i] * wr[i];
  logits[idx] = s;
}

// ---------------------------------------------------------------------------
// Per-row routing: softmax/tau, alpha-mix, shared floor, top-2, capacity.
// ---------------------------------------------------------------------------
__global__ __launch_bounds__(256)
void k_route(const float* __restrict__ logits, const int* __restrict__ t,
             float* __restrict__ disp, float* __restrict__ comb) {
  int row = blockIdx.x * 256 + threadIdx.x;
  if (row >= MROWS) return;
  int b = row >> 11;
  float tn = (float)t[b] / 1000.0f;
  float tau = 0.5f + 1.5f * tn;

  float lg[EE], p[EE];
  float m = -1e30f;
#pragma unroll
  for (int e = 0; e < EE; ++e) { lg[e] = logits[row * EE + e] / tau; m = fmaxf(m, lg[e]); }
  float sum = 0.0f;
#pragma unroll
  for (int e = 0; e < EE; ++e) { p[e] = __expf(lg[e] - m); sum += p[e]; }
  float inv = 1.0f / sum;
#pragma unroll
  for (int e = 0; e < EE; ++e) p[e] = 0.85f * (p[e] * inv) + 0.15f / 8.0f;

  float wmin = 0.1f + 0.1f * tn;
  float sh = fmaxf(p[0], wmin);
  float osum = 0.0f;
#pragma unroll
  for (int e = 1; e < EE; ++e) osum += p[e];
  float sc = (1.0f - sh) / fmaxf(osum, 1e-8f);
  p[0] = sh;
#pragma unroll
  for (int e = 1; e < EE; ++e) p[e] *= sc;

  // top-2 (ties -> lower index, matching jax.lax.top_k)
  int i1 = 0;
#pragma unroll
  for (int e = 1; e < EE; ++e) if (p[e] > p[i1]) i1 = e;
  int i2 = (i1 == 0) ? 1 : 0;
#pragma unroll
  for (int e = 0; e < EE; ++e) if (e != i1 && p[e] > p[i2]) i2 = e;

  float d[EE];
#pragma unroll
  for (int e = 0; e < EE; ++e) d[e] = 0.0f;
  d[i1] = p[i1]; d[i2] = p[i2];

  float cap = 0.5f + 0.1f * tn;
  float head[EE], exs = 0.0f, hs = 0.0f;
#pragma unroll
  for (int e = 0; e < EE; ++e) {
    float ex = fmaxf(d[e] - cap, 0.0f);
    float cp = d[e] - ex;
    head[e] = fmaxf(cap - cp, 0.0f);
    exs += ex; hs += head[e];
    d[e] = cp;
  }
  hs = fmaxf(hs, 1e-8f);
  float ds = 0.0f;
#pragma unroll
  for (int e = 0; e < EE; ++e) { d[e] += exs * head[e] / hs; ds += d[e]; }
  float cinv = 1.0f / (ds + 1e-8f);
#pragma unroll
  for (int e = 0; e < EE; ++e) {
    disp[row * EE + e] = d[e];
    comb[row * EE + e] = d[e] * cinv;
  }
}

// Deterministic single-block reduction for bias penalty
__global__ __launch_bounds__(256)
void k_penalty(const float* __restrict__ disp, float* __restrict__ out) {
  __shared__ float red[256];
  const int tid = threadIdx.x;
  float s[EE];
#pragma unroll
  for (int e = 0; e < EE; ++e) s[e] = 0.0f;
  for (int row = tid; row < MROWS; row += 256)
#pragma unroll
    for (int e = 0; e < EE; ++e) s[e] += disp[row * EE + e];
  float loadv[EE];
  for (int e = 0; e < EE; ++e) {
    red[tid] = s[e]; __syncthreads();
    for (int off = 128; off > 0; off >>= 1) {
      if (tid < off) red[tid] += red[tid + off];
      __syncthreads();
    }
    loadv[e] = red[0] * (1.0f / (float)BB);
    __syncthreads();
  }
  if (tid == 0) {
    float fair = (float)NBQ / 7.0f;
    float acc = 0.0f;
    for (int e = 1; e < EE; ++e) {
      float r = fmaxf(loadv[e] - 1.5f * fair, 0.0f);
      acc += r * r;
    }
    out[0] = 0.01f * acc;
  }
}

// ---------------------------------------------------------------------------
// Launch glue
// ---------------------------------------------------------------------------
extern "C" void kernel_launch(void* const* d_in, const int* in_sizes, int n_in,
                              void* d_out, int out_size, void* d_ws, size_t ws_size,
                              hipStream_t stream) {
  (void)in_sizes; (void)n_in; (void)out_size; (void)ws_size;
  const float* tokens = (const float*)d_in[0];
  const float* outA   = (const float*)d_in[1];
  const float* outC   = (const float*)d_in[2];
  const int*   t      = (const int*)d_in[3];
  const float* ipw    = (const float*)d_in[4];
  const float* ipb    = (const float*)d_in[5];
  const float* opw    = (const float*)d_in[6];
  const float* opb    = (const float*)d_in[7];
  const float* lng    = (const float*)d_in[8];
  const float* lnb    = (const float*)d_in[9];
  const float* tew1   = (const float*)d_in[10];
  const float* teb1   = (const float*)d_in[11];
  const float* tew2   = (const float*)d_in[12];
  const float* teb2   = (const float*)d_in[13];
  const float* gw1    = (const float*)d_in[14];
  const float* gb1    = (const float*)d_in[15];
  const float* gw2    = (const float*)d_in[16];
  const float* gb2    = (const float*)d_in[17];
  const float* ebias  = (const float*)d_in[18];

  char* ws = (char*)d_ws;
  size_t off = 0;
  auto carve = [&](size_t bytes) -> void* {
    void* p = ws + off;
    off += (bytes + 255) & ~(size_t)255;
    return p;
  };
  __bf16* tok_bf = (__bf16*)carve((size_t)MROWS * DD * 2);
  __bf16* ctx_bf = (__bf16*)carve((size_t)MROWS * DD * 2);
  __bf16* ipw_bf = (__bf16*)carve((size_t)3072 * DD * 2);
  __bf16* opw_bf = (__bf16*)carve((size_t)DD * DD * 2);
  __bf16* gw1_bf = (__bf16*)carve((size_t)DD * 3072 * 2);
  __bf16* Qb     = (__bf16*)carve((size_t)MROWS * DD * 2);
  __bf16* Kb     = (__bf16*)carve((size_t)MROWS * DD * 2);
  __bf16* Vb     = (__bf16*)carve((size_t)MROWS * DD * 2);
  __bf16* Ab     = (__bf16*)carve((size_t)MROWS * DD * 2);
  float*  projF  = (float*)carve((size_t)MROWS * DD * 4);
  float*  ctxB   = (float*)carve((size_t)MROWS * DD * 4);
  float*  embF   = (float*)carve((size_t)BB * DD * 4);
  float*  h1F    = (float*)carve((size_t)BB * 2048 * 4);
  float*  tembF  = (float*)carve((size_t)BB * DD * 4);
  __bf16* gin    = (__bf16*)carve((size_t)MROWS * 3072 * 2);
  __bf16* gh     = (__bf16*)carve((size_t)MROWS * DD * 2);
  float*  logits = (float*)carve((size_t)MROWS * EE * 4);

  // 1) precision conversion
  k_f2bf<<<2048, 256, 0, stream>>>(tokens, tok_bf, MROWS * DD);
  k_ctx_bf<<<2048, 256, 0, stream>>>(outA, outC, ctx_bf);
  k_f2bf<<<2048, 256, 0, stream>>>(ipw, ipw_bf, 3072 * DD);
  k_f2bf<<<1024, 256, 0, stream>>>(opw, opw_bf, DD * DD);
  k_f2bf<<<2048, 256, 0, stream>>>(gw1, gw1_bf, DD * 3072);

  // 2) Q/K/V projections (WMMA, async-DMA double-buffered)
  dim3 gemmGrid(DD / 128, MROWS / 128);
  k_gemm<<<gemmGrid, 256, 0, stream>>>(tok_bf, ipw_bf,             ipb,        nullptr, Qb, MROWS, DD, DD, 1);
  k_gemm<<<gemmGrid, 256, 0, stream>>>(ctx_bf, ipw_bf + 1024 * DD, ipb + 1024, nullptr, Kb, MROWS, DD, DD, 1);
  k_gemm<<<gemmGrid, 256, 0, stream>>>(ctx_bf, ipw_bf + 2048 * DD, ipb + 2048, nullptr, Vb, MROWS, DD, DD, 1);

  // 3) flash attention (WMMA)
  dim3 attnGrid(BB * HH, NBQ / 64);
  k_attn<<<attnGrid, 128, 0, stream>>>(Qb, Kb, Vb, Ab);

  // 4) out_proj + residual LayerNorm
  k_gemm<<<gemmGrid, 256, 0, stream>>>(Ab, opw_bf, opb, projF, nullptr, MROWS, DD, DD, 0);
  k_ln_residual<<<MROWS, 256, 0, stream>>>(projF, tokens, lng, lnb, ctxB);

  // 5) time-embedding MLP
  k_time_embed<<<BB, 512, 0, stream>>>(t, embF);
  k_te1<<<(BB * 2048) / 256, 256, 0, stream>>>(embF, tew1, teb1, h1F);
  k_te2<<<(BB * DD) / 256, 256, 0, stream>>>(h1F, tew2, teb2, tembF);

  // 6) gate network
  k_gate_in<<<8192, 256, 0, stream>>>(tokens, ctxB, tembF, gin, MROWS * 3072);
  k_gemm<<<gemmGrid, 256, 0, stream>>>(gin, gw1_bf, gb1, nullptr, gh, MROWS, DD, 3072, 2);
  k_gate2<<<(MROWS * EE) / 256, 256, 0, stream>>>(gh, gw2, gb2, ebias, logits);

  // 7) routing + penalty -> d_out = [dispatch | combine | penalty]
  float* disp = (float*)d_out;
  float* comb = disp + MROWS * EE;
  float* pen  = disp + 2 * MROWS * EE;
  k_route<<<MROWS / 256, 256, 0, stream>>>(logits, t, disp, comb);
  k_penalty<<<1, 256, 0, stream>>>(disp, pen);
}
